// convAttention_15564961481156
// MI455X (gfx1250) — compile-verified, matrix operand order checked by reading, not run on hardware
//
#include <hip/hip_runtime.h>
#include <hip/hip_bf16.h>

typedef __attribute__((ext_vector_type(16))) _Float16 v16h;
typedef __attribute__((ext_vector_type(8)))  _Float16 v8h;
typedef __attribute__((ext_vector_type(8)))  float    v8f;
typedef __attribute__((ext_vector_type(4)))  unsigned int tdm_u32x4;
typedef __attribute__((ext_vector_type(8)))  int          tdm_i32x8;
typedef __attribute__((ext_vector_type(4)))  int          tdm_i32x4;

#define NUM_HEADS 3
#define WIN  7
#define PAD  3
#define B_   8
#define C_   96
#define HD   32
#define H_   56
#define W_   56
#define HW   (H_ * W_)          /* 3136 */
#define BH   (B_ * NUM_HEADS)   /* 24   */
#define QKC  (2 * HD)           /* 64   */
#define TAPS (WIN * WIN)        /* 49   */
#define HALO_ROWS 9             /* <= 2 covered rows + 2*PAD + 1 */

// CDNA5 16-bit A/B fragment lane mapping (ISA 7.12.2):
// element e of lane l holds K = (e&7) + 8*(l>>4) + 16*(e>>3); row/col = l&15.
__device__ __forceinline__ int kmap(int lane, int e) {
  return (e & 7) + ((lane >> 4) << 3) + ((e >> 3) << 4);
}

__device__ __forceinline__ v16h cat8(v8h lo, v8h hi) {
  return __builtin_shufflevector(lo, hi, 0, 1, 2, 3, 4, 5, 6, 7,
                                 8, 9, 10, 11, 12, 13, 14, 15);
}

// Packed A-fragment: 16 contiguous halves per lane -> two b128 loads.
__device__ __forceinline__ v16h load_afrag(const _Float16* p) {
  v8h lo = *(const v8h*)p;
  v8h hi = *(const v8h*)(p + 8);
  return cat8(lo, hi);
}

// ---------------------------------------------------------------------------
// Prep 0: transpose x (B,C,HW) f32 -> xT (B,HW,C) f16 (K contiguous per pixel)
// ---------------------------------------------------------------------------
__global__ void transpose_x_kernel(const float* __restrict__ x, _Float16* __restrict__ xT) {
  const int idx = blockIdx.x * blockDim.x + threadIdx.x;
  if (idx >= B_ * C_ * HW) return;
  const int p = idx % HW;
  const int c = (idx / HW) % C_;
  const int b = idx / (C_ * HW);
  xT[((size_t)b * HW + p) * C_ + c] = (_Float16)x[idx];
}

// ---------------------------------------------------------------------------
// Prep 1: pack GEMM weights (M x K row-major f32) into per-lane-contiguous
// f16 fragments: out[((mtile*(K/32)+kchunk)*32 + lane)*16 + e]
// ---------------------------------------------------------------------------
__global__ void pack_w_frags_kernel(const float* __restrict__ W, _Float16* __restrict__ out,
                                    int M, int K) {
  const int idx = blockIdx.x * blockDim.x + threadIdx.x;
  const int total = (M / 16) * (K / 32) * 512;
  if (idx >= total) return;
  const int frag = idx >> 9;
  const int r = idx & 511;
  const int lane = r >> 4, e = r & 15;
  const int nk = K / 32;
  const int mtile = frag / nk, kchunk = frag % nk;
  const int m = mtile * 16 + (lane & 15);
  const int k = kchunk * 32 + kmap(lane, e);
  out[idx] = (_Float16)W[m * K + k];
}

// ---------------------------------------------------------------------------
// Prep 2: pack w_dm (49,64,7,7) f32 -> f16 fragments, 49 out-ch padded to 64.
// Layout: (((t*2 + chunk)*32 + lane)*4 + otile)*16 + e  -- a lane's 4 o-tile
// fragments are 128 contiguous bytes (clause of b128 loads).
// ---------------------------------------------------------------------------
__global__ void pack_dm_frags_kernel(const float* __restrict__ w_dm, _Float16* __restrict__ out) {
  const int idx = blockIdx.x * blockDim.x + threadIdx.x;
  const int total = TAPS * 2 * 2048;
  if (idx >= total) return;
  const int slot = idx / 2048;           // t*2 + chunk
  const int r = idx % 2048;
  const int lane = r / 64;
  const int r2 = r % 64;
  const int otile = r2 / 16, e = r2 % 16;
  const int t = slot >> 1, chunk = slot & 1;
  const int o = otile * 16 + (lane & 15);
  const int k = chunk * 32 + kmap(lane, e);
  const float v = (o < TAPS)
      ? w_dm[(((size_t)o * QKC + k) * WIN + t / WIN) * WIN + (t % WIN)]
      : 0.f;
  out[idx] = (_Float16)v;
}

// ---------------------------------------------------------------------------
// Kernel 1: QKV projection GEMM.  4 waves/block, each wave one 16-pixel tile.
// B operand: contiguous b128 loads from xT.  A: packed fragments.
// q,k routed (q pre-scaled) into pixel-major qk16[bh][p][64]; v kept f32 NCHW.
// ---------------------------------------------------------------------------
__global__ __launch_bounds__(128)
void qkv_gemm_kernel(const _Float16* __restrict__ xT, const _Float16* __restrict__ wfrag,
                     const float* __restrict__ b_qkv,
                     _Float16* __restrict__ qk16, float* __restrict__ vbuf) {
  const int lane = threadIdx.x;
  const int p0 = (blockIdx.x * 4 + threadIdx.y) * 16;
  const int m0 = blockIdx.y * 16;
  const int b  = blockIdx.z;
  const int n  = lane & 15;
  const int p  = p0 + n;
  const int koff = (lane >> 4) << 3;  // 0 or 8
  const _Float16* brec  = xT + ((size_t)b * HW + p) * C_ + koff;
  const _Float16* afrag = wfrag + ((size_t)blockIdx.y * 3 * 32 + lane) * 16;
  v8f acc = {};
#pragma unroll
  for (int kidx = 0; kidx < 3; ++kidx) {
    v16h a = load_afrag(afrag + kidx * 512);
    v8h blo = *(const v8h*)(brec + kidx * 32);
    v8h bhi = *(const v8h*)(brec + kidx * 32 + 16);
    v16h bf = cat8(blo, bhi);
    acc = __builtin_amdgcn_wmma_f32_16x16x32_f16(false, a, false, bf,
                                                 (short)0, acc, false, false);
  }
  const float scale = 0.17677669529663687f;  // 32^-0.5
#pragma unroll
  for (int f = 0; f < 8; ++f) {
    const int o = m0 + f + ((lane >> 4) << 3);
    const float val = acc[f] + b_qkv[o];
    const int s = o / C_;        // 0=q, 1=k, 2=v
    const int r = o % C_;
    const int head = r / HD;
    const int d = r % HD;
    if (s == 0) {
      qk16[(((size_t)b * NUM_HEADS + head) * HW + p) * QKC + d] = (_Float16)(val * scale);
    } else if (s == 1) {
      qk16[(((size_t)b * NUM_HEADS + head) * HW + p) * QKC + HD + d] = (_Float16)val;
    } else {
      vbuf[((size_t)b * C_ + r) * HW + p] = val;
    }
  }
}

// ---------------------------------------------------------------------------
// Kernel 2: attention logits via implicit-GEMM 7x7 conv.
// Block = 4 waves = 64 pixels of one bh.  The contiguous pixel-major halo
// ([pstart, pstart+nrow*56) x 64ch f16, <= 64.5 KB) is staged into LDS with
// one TDM tensor_load_to_lds, then each wave accumulates ALL 4 o-tiles:
// per tap = 16 b128 A loads (identical addrs across waves -> L0 broadcast)
//         + 4 predicated ds_load_b128 B loads + 8 WMMA.
// ---------------------------------------------------------------------------
__global__ __launch_bounds__(128)
void attn_conv_kernel(const _Float16* __restrict__ qk, const _Float16* __restrict__ wfrag,
                      const float* __restrict__ b_dm, const float* __restrict__ bias_table,
                      float* __restrict__ attn) {
  __shared__ _Float16 sqk[HALO_ROWS * W_ * QKC];  // 64512 B
  const int lane = threadIdx.x;
  const int bh = blockIdx.z;
  const int head = bh % NUM_HEADS;
  const int pblk = blockIdx.x * 64;
  const int y_first = pblk / W_;
  const int y_last  = (pblk + 63) / W_;
  const int ylo = (y_first - PAD < 0) ? 0 : (y_first - PAD);
  const int yhi = (y_last + PAD > H_ - 1) ? (H_ - 1) : (y_last + PAD);
  const int pstart = ylo * W_;
  const int nrow = yhi - ylo + 1;

  if (threadIdx.y == 0) {
    // --- Tensor Data Mover: 1-row tile of n8 8-byte elements (ISA 8.3-8.5) ---
    const unsigned lds_base = (unsigned)(unsigned long long)(const void*)sqk;
    const unsigned long long ga =
        (unsigned long long)(const void*)(qk + ((size_t)bh * HW + pstart) * QKC);
    const unsigned n8 = (unsigned)(nrow * W_ * QKC) / 4;  // halves*2/8
    tdm_u32x4 g0;
    g0.x = 1u;                                   // count=1, user descriptor
    g0.y = lds_base;                             // lds_addr (bytes)
    g0.z = (unsigned)(ga & 0xffffffffu);         // global_addr[31:0]
    g0.w = (unsigned)(ga >> 32) | 0x80000000u;   // global_addr[56:32] | type=2
    tdm_i32x8 g1;
    g1[0] = 0x30000;                             // data_size=3 (8B), mask=0
    g1[1] = (int)((n8 & 0xffffu) << 16);         // tensor_dim0[15:0]
    g1[2] = (int)((n8 >> 16) | (1u << 16));      // tensor_dim0[31:16] | tensor_dim1=1
    g1[3] = (int)(n8 << 16);                     // tile_dim0 = n8
    g1[4] = 1;                                   // tile_dim1 = 1
    g1[5] = (int)n8;                             // tensor_dim0_stride[31:0]
    g1[6] = 0;
    g1[7] = 0;
    tdm_i32x4 gz4 = {0, 0, 0, 0};
    tdm_i32x8 gz8 = {0, 0, 0, 0, 0, 0, 0, 0};
    __builtin_amdgcn_tensor_load_to_lds(g0, g1, gz4, gz4, gz8, 0);
    __builtin_amdgcn_s_wait_tensorcnt(0);
  }
  __syncthreads();

  const int p = pblk + threadIdx.y * 16 + (lane & 15);
  const int y = p / W_, x = p % W_;
  const int koff = (lane >> 4) << 3;  // 0 or 8
  v8f accs[4] = {};
  for (int t = 0; t < TAPS; ++t) {
    const int kh = t / WIN, kw = t % WIN;
    const int ys = y + kh - PAD, xs = x + kw - PAD;
    const bool inb = (ys >= 0) && (ys < H_) && (xs >= 0) && (xs < W_);
    const int lp = ys * W_ + xs - pstart;        // local pixel in LDS halo
    const _Float16* bl = sqk + (long long)lp * QKC + koff;
    const _Float16* ab0 = wfrag + (((size_t)t * 2) * 32 + lane) * 64;
    __builtin_prefetch(ab0 + 4096, 0, 1);        // next tap's weights
#pragma unroll
    for (int chunk = 0; chunk < 2; ++chunk) {
      v8h zero = {};
      v8h blo = inb ? *(const v8h*)(bl + chunk * 32)      : zero;
      v8h bhi = inb ? *(const v8h*)(bl + chunk * 32 + 16) : zero;
      v16h bf = cat8(blo, bhi);
      const _Float16* ab = ab0 + (size_t)chunk * 2048;
      v16h a0 = load_afrag(ab);
      v16h a1 = load_afrag(ab + 16);
      v16h a2 = load_afrag(ab + 32);
      v16h a3 = load_afrag(ab + 48);
      accs[0] = __builtin_amdgcn_wmma_f32_16x16x32_f16(false, a0, false, bf,
                                                       (short)0, accs[0], false, false);
      accs[1] = __builtin_amdgcn_wmma_f32_16x16x32_f16(false, a1, false, bf,
                                                       (short)0, accs[1], false, false);
      accs[2] = __builtin_amdgcn_wmma_f32_16x16x32_f16(false, a2, false, bf,
                                                       (short)0, accs[2], false, false);
      accs[3] = __builtin_amdgcn_wmma_f32_16x16x32_f16(false, a3, false, bf,
                                                       (short)0, accs[3], false, false);
    }
  }
#pragma unroll
  for (int ot = 0; ot < 4; ++ot) {
#pragma unroll
    for (int f = 0; f < 8; ++f) {
      const int o = ot * 16 + f + ((lane >> 4) << 3);
      if (o < TAPS) {
        attn[((size_t)bh * TAPS + o) * HW + p] =
            accs[ot][f] + b_dm[o] + bias_table[o * NUM_HEADS + head];
      }
    }
  }
}

// ---------------------------------------------------------------------------
// Kernel 3: softmax over the 49 taps (axis=2), in place.
// ---------------------------------------------------------------------------
__global__ void softmax_kernel(float* __restrict__ attn) {
  const int idx = blockIdx.x * blockDim.x + threadIdx.x;
  if (idx >= BH * HW) return;
  const int bh = idx / HW;
  const int p  = idx % HW;
  float v[TAPS];
  float mx = -3.402823466e38f;
#pragma unroll
  for (int i = 0; i < TAPS; ++i) {
    v[i] = attn[((size_t)bh * TAPS + i) * HW + p];
    mx = fmaxf(mx, v[i]);
  }
  float sum = 0.f;
#pragma unroll
  for (int i = 0; i < TAPS; ++i) {
    v[i] = __expf(v[i] - mx);
    sum += v[i];
  }
  const float inv = 1.f / sum;
#pragma unroll
  for (int i = 0; i < TAPS; ++i)
    attn[((size_t)bh * TAPS + i) * HW + p] = v[i] * inv;
}

// ---------------------------------------------------------------------------
// Kernel 4: attention apply (weighted 7x7 gather of v).  Small FLOPs -> VALU.
// Writes pixel-major f16 activations for the projection GEMM's B operand.
// ---------------------------------------------------------------------------
__global__ void apply_kernel(const float* __restrict__ attn, const float* __restrict__ vbuf,
                             _Float16* __restrict__ outp) {
  const int idx = blockIdx.x * blockDim.x + threadIdx.x;
  if (idx >= B_ * C_ * HW) return;
  const int p = idx % HW;
  const int c = (idx / HW) % C_;
  const int b = idx / (C_ * HW);
  const int head = c / HD;
  const int bh = b * NUM_HEADS + head;
  const int y = p / W_, x = p % W_;
  float acc = 0.f;
#pragma unroll
  for (int i = 0; i < TAPS; ++i) {
    const int ys = y + i / WIN - PAD;
    const int xs = x + i % WIN - PAD;
    if (ys >= 0 && ys < H_ && xs >= 0 && xs < W_) {
      acc += attn[((size_t)bh * TAPS + i) * HW + p] *
             vbuf[((size_t)b * C_ + c) * HW + ys * W_ + xs];
    }
  }
  outp[((size_t)b * HW + p) * C_ + c] = (_Float16)acc;
}

// ---------------------------------------------------------------------------
// Kernel 5: output projection GEMM (fp32 out), same structure as kernel 1.
// ---------------------------------------------------------------------------
__global__ __launch_bounds__(128)
void proj_gemm_kernel(const _Float16* __restrict__ outp, const _Float16* __restrict__ wfrag,
                      const float* __restrict__ b_proj, float* __restrict__ out) {
  const int lane = threadIdx.x;
  const int p0 = (blockIdx.x * 4 + threadIdx.y) * 16;
  const int m0 = blockIdx.y * 16;
  const int b  = blockIdx.z;
  const int n  = lane & 15;
  const int p  = p0 + n;
  const int koff = (lane >> 4) << 3;
  const _Float16* brec  = outp + ((size_t)b * HW + p) * C_ + koff;
  const _Float16* afrag = wfrag + ((size_t)blockIdx.y * 3 * 32 + lane) * 16;
  v8f acc = {};
#pragma unroll
  for (int kidx = 0; kidx < 3; ++kidx) {
    v16h a = load_afrag(afrag + kidx * 512);
    v8h blo = *(const v8h*)(brec + kidx * 32);
    v8h bhi = *(const v8h*)(brec + kidx * 32 + 16);
    v16h bf = cat8(blo, bhi);
    acc = __builtin_amdgcn_wmma_f32_16x16x32_f16(false, a, false, bf,
                                                 (short)0, acc, false, false);
  }
#pragma unroll
  for (int f = 0; f < 8; ++f) {
    const int o = m0 + f + ((lane >> 4) << 3);
    out[((size_t)b * C_ + o) * HW + p] = acc[f] + b_proj[o];
  }
}

// ---------------------------------------------------------------------------
extern "C" void kernel_launch(void* const* d_in, const int* in_sizes, int n_in,
                              void* d_out, int out_size, void* d_ws, size_t ws_size,
                              hipStream_t stream) {
  (void)in_sizes; (void)n_in; (void)out_size; (void)ws_size;
  const float* x      = (const float*)d_in[0];
  const float* w_qkv  = (const float*)d_in[1];
  const float* b_qkv  = (const float*)d_in[2];
  const float* w_dm   = (const float*)d_in[3];
  const float* b_dm   = (const float*)d_in[4];
  const float* bias_t = (const float*)d_in[5];
  const float* w_proj = (const float*)d_in[6];
  const float* b_proj = (const float*)d_in[7];
  float* out = (float*)d_out;

  char* ws = (char*)d_ws;
  size_t off = 0;
  auto alloc = [&](size_t bytes) {
    void* p = ws + off;
    off = (off + bytes + 255) & ~(size_t)255;
    return p;
  };
  _Float16* qk16   = (_Float16*)alloc((size_t)BH * HW * QKC * sizeof(_Float16));
  float*    vbuf   = (float*)   alloc((size_t)B_ * C_ * HW * sizeof(float));
  float*    attn   = (float*)   alloc((size_t)BH * TAPS * HW * sizeof(float));
  _Float16* outp   = (_Float16*)alloc((size_t)B_ * HW * C_ * sizeof(_Float16));
  _Float16* xT     = (_Float16*)alloc((size_t)B_ * HW * C_ * sizeof(_Float16));
  _Float16* wqkv_f = (_Float16*)alloc((size_t)18 * 3 * 512 * sizeof(_Float16));
  _Float16* wprj_f = (_Float16*)alloc((size_t)6 * 3 * 512 * sizeof(_Float16));
  _Float16* wdm_f  = (_Float16*)alloc((size_t)TAPS * 2 * 2048 * sizeof(_Float16));

  // --- prep ---
  transpose_x_kernel<<<dim3((B_ * C_ * HW + 255) / 256), dim3(256), 0, stream>>>(x, xT);
  pack_w_frags_kernel<<<dim3((18 * 3 * 512 + 255) / 256), dim3(256), 0, stream>>>(
      w_qkv, wqkv_f, 3 * C_, C_);
  pack_w_frags_kernel<<<dim3((6 * 3 * 512 + 255) / 256), dim3(256), 0, stream>>>(
      w_proj, wprj_f, C_, C_);
  pack_dm_frags_kernel<<<dim3((TAPS * 2 * 2048 + 255) / 256), dim3(256), 0, stream>>>(
      w_dm, wdm_f);

  // --- pipeline ---
  const dim3 wblock(32, 4);
  qkv_gemm_kernel<<<dim3(HW / 64, 18, B_), wblock, 0, stream>>>(
      xT, wqkv_f, b_qkv, qk16, vbuf);
  attn_conv_kernel<<<dim3(HW / 64, 1, BH), wblock, 0, stream>>>(
      qk16, wdm_f, b_dm, bias_t, attn);
  softmax_kernel<<<dim3((BH * HW + 255) / 256), dim3(256), 0, stream>>>(attn);
  apply_kernel<<<dim3((B_ * C_ * HW + 255) / 256), dim3(256), 0, stream>>>(
      attn, vbuf, outp);
  proj_gemm_kernel<<<dim3(HW / 64, C_ / 16, B_), wblock, 0, stream>>>(
      outp, wprj_f, b_proj, out);
}
